// EdgeColoringGNN_36550171689550
// MI455X (gfx1250) — matrix operand order repaired and checked by reading, not compiled, not run on hardware
//
#include <hip/hip_runtime.h>

typedef _Float16 f16_t;
typedef __attribute__((ext_vector_type(16))) _Float16 v16h;
typedef __attribute__((ext_vector_type(8)))  float    v8f;

#define N_NODES 50000
#define N_EDGES 1600000
#define FN 8
#define FE 8
#define H  64
#define NL 3
#define NC 10
#define KPAD 160        // 2H + FE = 136 padded to 5*32

// packed fragment sizes (in f16 elements)
#define W1P_ELEMS (4*5*32*16)   // 10240
#define W2P_ELEMS (2*2*32*16)   // 2048
#define W3P_ELEMS (1*1*32*16)   // 512
#define WCP_ELEMS (3*4*2*32*16) // 12288
#define PACK_TOTAL (W1P_ELEMS + W2P_ELEMS + W3P_ELEMS + WCP_ELEMS)

static __device__ __forceinline__ v8f wmma32(v16h a, v16h b, v8f c) {
  // emits v_wmma_f32_16x16x32_f16
  return __builtin_amdgcn_wmma_f32_16x16x32_f16(false, a, false, b, (short)0, c,
                                                false, false);
}

// A-fragment (16x32 f16) from an LDS row-major tile with given row stride.
// ISA layout: lanes 0-15: M=lane, K = kc*32 + {0..7, 16..23}
//             lanes 16-31: M=lane-16, K = kc*32 + {8..15, 24..31}
static __device__ __forceinline__ v16h fragA_lds(const f16_t* base, int stride,
                                                 int lane, int kc) {
  int m = lane & 15;
  const f16_t* p = base + m * stride + kc * 32 + ((lane < 16) ? 0 : 8);
  v16h a;
#pragma unroll
  for (int i = 0; i < 8; ++i) { a[i] = p[i]; a[i + 8] = p[i + 16]; }
  return a;
}

// A-fragment built from a global f32 row-major tile (stride 64), converted f16.
static __device__ __forceinline__ v16h fragA_g32(const float* base, int lane, int kc) {
  int m = lane & 15;
  const float* p = base + m * H + kc * 32 + ((lane < 16) ? 0 : 8);
  v16h a;
#pragma unroll
  for (int i = 0; i < 8; ++i) {
    a[i]     = (f16_t)p[i];
    a[i + 8] = (f16_t)p[i + 16];
  }
  return a;
}

// ---------------- weight packing into B-fragment layout ----------------
// B layout (dense 32x16 f16, 8 VGPRs): lanes 0-15 hold K={0..7,16..23},
// lanes 16-31 hold K={8..15,24..31}; N = lane&15. Stored [frag][lane][16] contiguous.
__global__ void k_pack_weights(const float* __restrict__ W1,
                               const float* __restrict__ W2,
                               const float* __restrict__ W3,
                               const float* __restrict__ Wc,
                               f16_t* __restrict__ W1p, f16_t* __restrict__ W2p,
                               f16_t* __restrict__ W3p, f16_t* __restrict__ Wcp) {
  int idx = blockIdx.x * blockDim.x + threadIdx.x;
  if (idx >= PACK_TOTAL) return;
  if (idx < W1P_ELEMS) {
    int i = idx & 15, lane = (idx >> 4) & 31, frag = idx >> 9;
    int kc = frag % 5, nt = frag / 5;
    int k = kc * 32 + ((lane < 16) ? 0 : 8) + ((i < 8) ? i : i + 8);
    int n = nt * 16 + (lane & 15);
    W1p[idx] = (f16_t)((k < (2 * H + FE)) ? W1[k * H + n] : 0.0f);
  } else if (idx < W1P_ELEMS + W2P_ELEMS) {
    int j = idx - W1P_ELEMS;
    int i = j & 15, lane = (j >> 4) & 31, frag = j >> 9;
    int kc = frag & 1, nt = frag >> 1;
    int k = kc * 32 + ((lane < 16) ? 0 : 8) + ((i < 8) ? i : i + 8);
    int n = nt * 16 + (lane & 15);
    W2p[j] = (f16_t)W2[k * (H / 2) + n];
  } else if (idx < W1P_ELEMS + W2P_ELEMS + W3P_ELEMS) {
    int j = idx - (W1P_ELEMS + W2P_ELEMS);
    int i = j & 15, lane = (j >> 4) & 31;
    int k = ((lane < 16) ? 0 : 8) + ((i < 8) ? i : i + 8);
    int n = lane & 15;
    W3p[j] = (f16_t)((n < NC) ? W3[k * NC + n] : 0.0f);
  } else {
    int j = idx - (W1P_ELEMS + W2P_ELEMS + W3P_ELEMS);
    int i = j & 15, lane = (j >> 4) & 31, frag = j >> 9;
    int l = frag >> 3, r8 = frag & 7, nt = r8 >> 1, kc = r8 & 1;
    int k = kc * 32 + ((lane < 16) ? 0 : 8) + ((i < 8) ? i : i + 8);
    int n = nt * 16 + (lane & 15);
    Wcp[j] = (f16_t)Wc[(l * H + k) * H + n];
  }
}

// ---------------- degree / normalization ----------------
__global__ void k_deg_init(float* __restrict__ deg) {
  int n = blockIdx.x * blockDim.x + threadIdx.x;
  if (n < N_NODES) deg[n] = 1.0f;  // self loop
}
__global__ void k_deg_count(const int* __restrict__ dst, float* __restrict__ deg) {
  int e = blockIdx.x * blockDim.x + threadIdx.x;
  if (e < N_EDGES) atomicAdd(&deg[dst[e]], 1.0f);
}
__global__ void k_deg_fin(float* __restrict__ deg) {
  int n = blockIdx.x * blockDim.x + threadIdx.x;
  if (n < N_NODES) deg[n] = rsqrtf(deg[n]);  // deg >= 1 always
}

// ---------------- node encoder: h = x @ W_node + b_node ----------------
__global__ void k_node_enc(const float* __restrict__ x, const float* __restrict__ Wn,
                           const float* __restrict__ bn, float* __restrict__ h) {
  int t = blockIdx.x * blockDim.x + threadIdx.x;
  if (t >= N_NODES * H) return;
  int n = t >> 6, j = t & 63;
  float acc = bn[j];
#pragma unroll
  for (int k = 0; k < FN; ++k) acc += x[n * FN + k] * Wn[k * H + j];
  h[t] = acc;
}

// ---------------- hw = h @ W_conv[l]  (WMMA, one wave per 16-node tile) --------
__global__ void k_gemm_h(const float* __restrict__ hin,
                         const f16_t* __restrict__ Wcl,  // 8 fragments
                         float* __restrict__ hout) {
  int wv = threadIdx.x >> 5, lane = threadIdx.x & 31;
  int tile = blockIdx.x * 4 + wv;
  if (tile >= N_NODES / 16) return;
  const float* tbase = hin + (size_t)tile * 16 * H;
  v16h a0 = fragA_g32(tbase, lane, 0);
  v16h a1 = fragA_g32(tbase, lane, 1);
  int mbase = (lane < 16) ? 0 : 8;
  int nlo = lane & 15;
#pragma unroll
  for (int nt = 0; nt < 4; ++nt) {
    v8f c = {};
    c = wmma32(a0, *(const v16h*)(Wcl + (nt * 2 + 0) * 512 + lane * 16), c);
    c = wmma32(a1, *(const v16h*)(Wcl + (nt * 2 + 1) * 512 + lane * 16), c);
    int n = nt * 16 + nlo;
#pragma unroll
    for (int r = 0; r < 8; ++r)
      hout[(size_t)(tile * 16 + mbase + r) * H + n] = c[r];
  }
}

// ---------------- acc init: self-loop term + bias ----------------
__global__ void k_acc_init(const float* __restrict__ hw, const float* __restrict__ dinv,
                           const float* __restrict__ bc, float* __restrict__ acc) {
  int t = blockIdx.x * blockDim.x + threadIdx.x;
  if (t >= N_NODES * H) return;
  int n = t >> 6, j = t & 63;
  float di = dinv[n];
  acc[t] = bc[j] + hw[t] * di * di;
}

// ---------------- scatter: acc[dst] += hw[src]*dinv[src]*dinv[dst] ----------------
__global__ void k_scatter(const int* __restrict__ src, const int* __restrict__ dst,
                          const float* __restrict__ hw, const float* __restrict__ dinv,
                          float* __restrict__ acc) {
  int t = blockIdx.x * blockDim.x + threadIdx.x;
  if (t >= N_EDGES * 64) return;
  int e = t >> 6, j = t & 63;
  int s = src[e], d = dst[e];
  float v = hw[s * H + j] * dinv[s] * dinv[d];
  atomicAdd(&acc[d * H + j], v);
}

__global__ void k_relu(float* __restrict__ h) {
  int t = blockIdx.x * blockDim.x + threadIdx.x;
  if (t < N_NODES * H) h[t] = fmaxf(h[t], 0.0f);
}

// last layer: relu + convert to f16 (halves edge-MLP gather bandwidth)
__global__ void k_relu_cvt(const float* __restrict__ h, f16_t* __restrict__ h16) {
  int t = blockIdx.x * blockDim.x + threadIdx.x;
  if (t < N_NODES * H) h16[t] = (f16_t)fmaxf(h[t], 0.0f);
}

// ---------------- edge MLP: [h[src],h[dst],eattr] -> 64 -> 32 -> 10 ----------------
__global__ void __launch_bounds__(128)
k_edge_mlp(const f16_t* __restrict__ h16, const int* __restrict__ src,
           const int* __restrict__ dst, const float* __restrict__ eattr,
           const f16_t* __restrict__ W1p, const f16_t* __restrict__ W2p,
           const f16_t* __restrict__ W3p, const float* __restrict__ b1,
           const float* __restrict__ b2, const float* __restrict__ b3,
           float* __restrict__ out) {
  __shared__ f16_t As[4][16 * KPAD];   // 20 KB
  __shared__ f16_t Z1[4][16 * H];      //  8 KB
  __shared__ f16_t Z2[4][16 * (H/2)];  //  4 KB

  int wv = threadIdx.x >> 5, lane = threadIdx.x & 31;
  int tile = blockIdx.x * 4 + wv;
  const int ntiles = N_EDGES / 16;
  bool valid = tile < ntiles;
  if (!valid) tile = ntiles - 1;  // keep barriers uniform
  int e0 = tile * 16;

  // ---- stage A tile h-parts with async global->LDS copies ----
  // 16 rows x (8 src-chunks + 8 dst-chunks) of 16 bytes; 32 lanes -> 8 iters.
#pragma unroll
  for (int it = 0; it < 8; ++it) {
    int id = it * 32 + lane;          // 0..255
    int row = id >> 4;
    int c = id & 15;
    int e = e0 + row;
    int node = (c < 8) ? src[e] : dst[e];
    const f16_t* g = h16 + (size_t)node * H + (c & 7) * 8;
    unsigned laddr = (unsigned)(uintptr_t)&As[wv][row * KPAD + c * 8];
    asm volatile("global_load_async_to_lds_b128 %0, %1, off"
                 :: "v"(laddr), "v"(g) : "memory");
  }
  // attr (f32 -> f16) and zero padding via normal LDS stores
  if (lane < 16) {
    int e = e0 + lane;
    const float* ea = eattr + (size_t)e * FE;
    f16_t* arow = &As[wv][lane * KPAD];
#pragma unroll
    for (int k = 0; k < FE; ++k) arow[128 + k] = (f16_t)ea[k];
  } else {
    f16_t* arow = &As[wv][(lane - 16) * KPAD];
#pragma unroll
    for (int k = 136; k < KPAD; ++k) arow[k] = (f16_t)0.0f;
  }
  asm volatile("s_wait_asynccnt 0x0" ::: "memory");
  __syncthreads();

  int mbase = (lane < 16) ? 0 : 8;
  int nlo = lane & 15;

  // ---- z1 = relu(A @ W1 + b1): 4 N-tiles x 5 K-chunks ----
#pragma unroll
  for (int nt = 0; nt < 4; ++nt) {
    v8f c = {};
#pragma unroll
    for (int kc = 0; kc < 5; ++kc) {
      v16h a = fragA_lds(&As[wv][0], KPAD, lane, kc);
      v16h b = *(const v16h*)(W1p + (nt * 5 + kc) * 512 + lane * 16);
      c = wmma32(a, b, c);
    }
    int n = nt * 16 + nlo;
    float bias = b1[n];
#pragma unroll
    for (int r = 0; r < 8; ++r) {
      float v = fmaxf(c[r] + bias, 0.0f);
      Z1[wv][(mbase + r) * H + n] = (f16_t)v;
    }
  }
  __syncthreads();

  // ---- z2 = relu(z1 @ W2 + b2): 2 N-tiles x 2 K-chunks ----
#pragma unroll
  for (int nt = 0; nt < 2; ++nt) {
    v8f c = {};
#pragma unroll
    for (int kc = 0; kc < 2; ++kc) {
      v16h a = fragA_lds(&Z1[wv][0], H, lane, kc);
      v16h b = *(const v16h*)(W2p + (nt * 2 + kc) * 512 + lane * 16);
      c = wmma32(a, b, c);
    }
    int n = nt * 16 + nlo;
    float bias = b2[n];
#pragma unroll
    for (int r = 0; r < 8; ++r) {
      float v = fmaxf(c[r] + bias, 0.0f);
      Z2[wv][(mbase + r) * (H / 2) + n] = (f16_t)v;
    }
  }
  __syncthreads();

  // ---- out = z2 @ W3 + b3: 1 tile, K=32, 10 valid columns ----
  {
    v8f c = {};
    v16h a = fragA_lds(&Z2[wv][0], H / 2, lane, 0);
    v16h b = *(const v16h*)(W3p + lane * 16);
    c = wmma32(a, b, c);
    if (valid && nlo < NC) {
      float bias = b3[nlo];
#pragma unroll
      for (int r = 0; r < 8; ++r) {
        int e = e0 + mbase + r;
        out[(size_t)e * NC + nlo] = c[r] + bias;
      }
    }
  }
}

extern "C" void kernel_launch(void* const* d_in, const int* in_sizes, int n_in,
                              void* d_out, int out_size, void* d_ws, size_t ws_size,
                              hipStream_t stream) {
  (void)in_sizes; (void)n_in; (void)out_size; (void)ws_size;
  const float* x     = (const float*)d_in[0];
  const int*   ei    = (const int*)d_in[1];
  const float* eattr = (const float*)d_in[2];
  const float* Wn    = (const float*)d_in[3];
  const float* bn    = (const float*)d_in[4];
  const float* Wc    = (const float*)d_in[5];
  const float* bc    = (const float*)d_in[6];
  const float* W1    = (const float*)d_in[7];
  const float* b1    = (const float*)d_in[8];
  const float* W2    = (const float*)d_in[9];
  const float* b2    = (const float*)d_in[10];
  const float* W3    = (const float*)d_in[11];
  const float* b3    = (const float*)d_in[12];
  float* out = (float*)d_out;
  const int* src = ei;
  const int* dst = ei + N_EDGES;

  // workspace layout (~32.3 MB, all sub-buffers 32B aligned)
  float* hX   = (float*)d_ws;
  float* hY   = hX + (size_t)N_NODES * H;
  float* dinv = hY + (size_t)N_NODES * H;
  f16_t* W1p  = (f16_t*)(dinv + N_NODES);
  f16_t* W2p  = W1p + W1P_ELEMS;
  f16_t* W3p  = W2p + W2P_ELEMS;
  f16_t* Wcp  = W3p + W3P_ELEMS;
  f16_t* h16  = Wcp + WCP_ELEMS;

  const int TPB = 256;
  // pack weights into WMMA B-fragment layout
  k_pack_weights<<<(PACK_TOTAL + TPB - 1) / TPB, TPB, 0, stream>>>(
      W1, W2, W3, Wc, W1p, W2p, W3p, Wcp);

  // degree / symmetric normalization (deg computed into dinv buffer in place)
  k_deg_init<<<(N_NODES + TPB - 1) / TPB, TPB, 0, stream>>>(dinv);
  k_deg_count<<<(N_EDGES + TPB - 1) / TPB, TPB, 0, stream>>>(dst, dinv);
  k_deg_fin<<<(N_NODES + TPB - 1) / TPB, TPB, 0, stream>>>(dinv);

  // node encoder -> hX
  k_node_enc<<<(N_NODES * H + TPB - 1) / TPB, TPB, 0, stream>>>(x, Wn, bn, hX);

  // 3 GCN layers: hX -> hY (GEMM) -> hX (init + scatter + relu)
  for (int l = 0; l < NL; ++l) {
    k_gemm_h<<<((N_NODES / 16) + 3) / 4, 128, 0, stream>>>(hX, Wcp + l * 8 * 512, hY);
    k_acc_init<<<(N_NODES * H + TPB - 1) / TPB, TPB, 0, stream>>>(hY, dinv, bc + l * H, hX);
    k_scatter<<<(N_EDGES * 64 + TPB - 1) / TPB, TPB, 0, stream>>>(src, dst, hY, dinv, hX);
    if (l < NL - 1)
      k_relu<<<(N_NODES * H + TPB - 1) / TPB, TPB, 0, stream>>>(hX);
    else
      k_relu_cvt<<<(N_NODES * H + TPB - 1) / TPB, TPB, 0, stream>>>(hX, h16);
  }

  // edge MLP with WMMA; activations gathered as f16 via async global->LDS DMA
  k_edge_mlp<<<(N_EDGES / 16) / 4, 128, 0, stream>>>(
      h16, src, dst, eattr, W1p, W2p, W3p, b1, b2, b3, out);
}
